// MultiHeadAttention_23905787970213
// MI455X (gfx1250) — matrix operand order repaired and checked by reading, not compiled, not run on hardware
//
#include <hip/hip_runtime.h>
#include <stdint.h>

// MI455X / gfx1250, wave32, WMMA f32_16x16x32_f16 throughout.
// Async global->LDS staging (ASYNCcnt path) when the toolchain exposes it.

typedef __attribute__((ext_vector_type(16))) _Float16 v16h;
typedef __attribute__((ext_vector_type(8)))  _Float16 v8h;
typedef __attribute__((ext_vector_type(8)))  float    v8f;
typedef __attribute__((ext_vector_type(4)))  float    v4f;
typedef int v4i __attribute__((__vector_size__(4 * sizeof(int))));

static constexpr int NB     = 2;     // batch
static constexpr int SEQ    = 2048;  // sequence length
static constexpr int DMODEL = 1024;  // hidden
static constexpr int NH     = 16;    // heads
static constexpr int DHEAD  = 64;    // head dim
static constexpr int EMB    = NH * DHEAD;  // 1024

#define DEV __device__ __forceinline__

#if defined(__has_builtin)
#  if __has_builtin(__builtin_amdgcn_global_load_async_to_lds_b128)
#    define HAVE_ASYNC 1
#  endif
#endif
#ifndef HAVE_ASYNC
#  define HAVE_ASYNC 0
#endif

#if HAVE_ASYNC
typedef __attribute__((address_space(1))) v4i gv4i;  // global int4
typedef __attribute__((address_space(3))) v4i lv4i;  // LDS int4
#endif

DEV v8f wmma16(v16h a, v16h b, v8f c) {
  // D = A(16x32 f16) * B(32x16 f16) + C(16x16 f32)
  return __builtin_amdgcn_wmma_f32_16x16x32_f16(false, a, false, b, (short)0, c,
                                                false, false);
}

// 16-byte global->LDS copy; async (GLOBAL_LOAD_ASYNC_TO_LDS_B128) if available.
DEV void copy16(const _Float16* g, _Float16* l) {
#if HAVE_ASYNC
  __builtin_amdgcn_global_load_async_to_lds_b128(
      (gv4i*)(uintptr_t)g, (lv4i*)(uintptr_t)l, 0, 0);
#else
  *(v8h*)l = *(const v8h*)g;
#endif
}

DEV void copy_wait() {
#if HAVE_ASYNC
#  if __has_builtin(__builtin_amdgcn_s_wait_asynccnt)
  __builtin_amdgcn_s_wait_asynccnt(0);
#  else
  asm volatile("s_wait_asynccnt 0x0" ::: "memory");
#  endif
#endif
}

// A-fragment: lane holds row M=lane&15; K subsets {0-7,16-23} (lanes<16) or
// {8-15,24-31} (lanes>=16). Two aligned 16B loads from a row-major row.
DEV v16h frag_a(const _Float16* row, int lane) {
  const int kb = (lane & 16) ? 8 : 0;
  v8h lo = *(const v8h*)(row + kb);
  v8h hi = *(const v8h*)(row + kb + 16);
  v16h a;
#pragma unroll
  for (int i = 0; i < 8; ++i) { a[i] = lo[i]; a[i + 8] = hi[i]; }
  return a;
}

// B-fragment from a row-major B^T tile: lane holds col N=lane&15; K subsets
// {0-7,16-23} (lanes<16) or {8-15,24-31} (lanes>=16): two aligned 16B loads.
DEV v16h frag_b(const _Float16* row, int lane) {
  const int kb = (lane & 16) ? 8 : 0;
  v8h lo = *(const v8h*)(row + kb);
  v8h hi = *(const v8h*)(row + kb + 16);
  v16h b;
#pragma unroll
  for (int i = 0; i < 8; ++i) { b[i] = lo[i]; b[i + 8] = hi[i]; }
  return b;
}

// ---------------------------------------------------------------------------
// Kernel 1: fused QKV projection.  y = x @ W^T + b  (M=B*S=4096, N=1024, K=1024)
// blockIdx.z selects q/k/v.  q,k -> ws [B,H,S,DH] f16;  v -> ws [B,H,DH,S] f16.
// ---------------------------------------------------------------------------
__global__ __launch_bounds__(256) void qkv_proj_kernel(
    const float* __restrict__ Q, const float* __restrict__ K,
    const float* __restrict__ V, const float* __restrict__ Wq,
    const float* __restrict__ Wk, const float* __restrict__ Wv,
    const float* __restrict__ bq, const float* __restrict__ bk,
    const float* __restrict__ bv, _Float16* __restrict__ qws,
    _Float16* __restrict__ kws, _Float16* __restrict__ vtws) {
  __shared__ alignas(32) _Float16 Xs[64][32];
  __shared__ alignas(32) _Float16 Ws[64][32];
  const int tid = threadIdx.x, lane = tid & 31, wid = tid >> 5;
  const int mode = blockIdx.z;
  const float* X    = (mode == 0) ? Q  : (mode == 1) ? K  : V;
  const float* W    = (mode == 0) ? Wq : (mode == 1) ? Wk : Wv;
  const float* bias = (mode == 0) ? bq : (mode == 1) ? bk : bv;
  const int mBase = blockIdx.x * 64;
  const int nBase = blockIdx.y * 64;
  const int mt  = wid >> 1;        // M-tile 0..3
  const int nt0 = (wid & 1) * 2;   // first of two N-tiles
  v8f c0 = {}, c1 = {};
  const int lr = tid >> 2;         // coop-load row 0..63
  const int lc = (tid & 3) * 8;    // coop-load col 0,8,16,24

  for (int kt = 0; kt < DMODEL / 32; ++kt) {
    const int kBase = kt * 32;
    {
      const float* src = X + (size_t)(mBase + lr) * DMODEL + kBase + lc;
      v4f f0 = *(const v4f*)src;
      v4f f1 = *(const v4f*)(src + 4);
      _Float16* dst = &Xs[lr][lc];
#pragma unroll
      for (int i = 0; i < 4; ++i) { dst[i] = (_Float16)f0[i]; dst[i + 4] = (_Float16)f1[i]; }
      const float* wsrc = W + (size_t)(nBase + lr) * DMODEL + kBase + lc;
      v4f g0 = *(const v4f*)wsrc;
      v4f g1 = *(const v4f*)(wsrc + 4);
      _Float16* wdst = &Ws[lr][lc];
#pragma unroll
      for (int i = 0; i < 4; ++i) { wdst[i] = (_Float16)g0[i]; wdst[i + 4] = (_Float16)g1[i]; }
      if (kt + 1 < DMODEL / 32) __builtin_prefetch(src + 32, 0, 3);  // global_prefetch_b8
    }
    __syncthreads();
    v16h a  = frag_a(&Xs[mt * 16 + (lane & 15)][0], lane);
    v16h b0 = frag_b(&Ws[nt0 * 16 + (lane & 15)][0], lane);
    v16h b1 = frag_b(&Ws[(nt0 + 1) * 16 + (lane & 15)][0], lane);
    c0 = wmma16(a, b0, c0);
    c1 = wmma16(a, b1, c1);
    __syncthreads();
  }

#pragma unroll
  for (int t = 0; t < 2; ++t) {
    v8f c = t ? c1 : c0;
    const int nt = nt0 + t;
#pragma unroll
    for (int r = 0; r < 8; ++r) {
      const int gr = mBase + mt * 16 + r + ((lane & 16) ? 8 : 0);  // row in B*S
      const int gc = nBase + nt * 16 + (lane & 15);                // col in E
      const float val = c[r] + bias[gc];
      const int b = gr >> 11, s = gr & (SEQ - 1);
      const int h = gc >> 6, d = gc & 63;
      if (mode == 2)
        vtws[((size_t)(b * NH + h) * DHEAD + d) * SEQ + s] = (_Float16)val;
      else {
        _Float16* dst = (mode == 0) ? qws : kws;
        dst[((size_t)(b * NH + h) * SEQ + s) * DHEAD + d] = (_Float16)val;
      }
    }
  }
}

// ---------------------------------------------------------------------------
// Kernel 2: causal attention.  One block = 128 query rows of one (b,h).
// 8 waves, each owning a 16-row WMMA M-tile (per-row softmax state stays
// within half-wave lane groups).  Phase 1: online row max/sum via WMMA scores.
// Phase 2: recompute scores, write normalized probs (f32), and run P@V WMMAs
// from an LDS-staged f16 P tile.  Fully-masked columns are zero-filled.
// K / V^T tiles are staged with async global->LDS copies when available.
// ---------------------------------------------------------------------------
__global__ __launch_bounds__(256) void attn_kernel(
    const _Float16* __restrict__ qws, const _Float16* __restrict__ kws,
    const _Float16* __restrict__ vtws, _Float16* __restrict__ ctx,
    float* __restrict__ probs) {
  __shared__ alignas(32) _Float16 Ks[64][64];       // keys  (row = key, col = d)
  __shared__ alignas(32) _Float16 VTs[64][64];      // V^T   (row = d, col = key)
  __shared__ alignas(32) _Float16 Ps[8][16][64];    // per-wave P tile
  const int tid = threadIdx.x, lane = tid & 31, wid = tid >> 5;
  const int qb = blockIdx.x;            // query block (128 rows)
  const int bh = blockIdx.y;            // b*NH + h
  const int mb = qb * 128 + wid * 16;   // this wave's first query row
  const int rbase = mb + ((lane & 16) ? 8 : 0);
  const int njb = 2 * qb + 2;           // causal key-block count (64-wide)
  const float scl = 0.125f;             // 1/sqrt(64)

  const _Float16* qrow = qws + ((size_t)bh * SEQ + mb + (lane & 15)) * DHEAD;
  const v16h aq0 = frag_a(qrow, lane);
  const v16h aq1 = frag_a(qrow + 32, lane);

  const int lr = tid >> 2;        // coop-load row 0..63
  const int lc = (tid & 3) * 16;  // coop-load col 0,16,32,48

  float m[8], l[8];
#pragma unroll
  for (int r = 0; r < 8; ++r) { m[r] = -3.0e38f; l[r] = 0.0f; }

  // -------- phase 1: row max & exp-sum --------
  for (int jb = 0; jb < njb; ++jb) {
    {
      const _Float16* ksrc =
          kws + ((size_t)bh * SEQ + jb * 64 + lr) * DHEAD + lc;
      copy16(ksrc, &Ks[lr][lc]);
      copy16(ksrc + 8, &Ks[lr][lc + 8]);
    }
    copy_wait();
    __syncthreads();
#pragma unroll
    for (int nt = 0; nt < 4; ++nt) {
      v16h b0 = frag_b(&Ks[nt * 16 + (lane & 15)][0], lane);
      v16h b1 = frag_b(&Ks[nt * 16 + (lane & 15)][32], lane);
      v8f c = {};
      c = wmma16(aq0, b0, c);
      c = wmma16(aq1, b1, c);
      const int key = jb * 64 + nt * 16 + (lane & 15);
#pragma unroll
      for (int r = 0; r < 8; ++r) {
        float s = c[r] * scl;
        if (key > rbase + r) s = -1.0e9f;  // causal mask
        float tm = s;
#pragma unroll
        for (int off = 8; off; off >>= 1) tm = fmaxf(tm, __shfl_xor(tm, off, 32));
        const float mn = fmaxf(m[r], tm);
        float e = __expf(s - mn);
#pragma unroll
        for (int off = 8; off; off >>= 1) e += __shfl_xor(e, off, 32);
        l[r] = l[r] * __expf(m[r] - mn) + e;
        m[r] = mn;
      }
    }
    __syncthreads();
  }
  float linv[8];
#pragma unroll
  for (int r = 0; r < 8; ++r) linv[r] = 1.0f / l[r];

  // -------- phase 2: probs out + context accumulation --------
  v8f a0 = {}, a1 = {}, a2 = {}, a3 = {};
  const size_t pbase = (size_t)bh * SEQ * SEQ;
  for (int jb = 0; jb < njb; ++jb) {
    {
      const _Float16* ksrc =
          kws + ((size_t)bh * SEQ + jb * 64 + lr) * DHEAD + lc;
      copy16(ksrc, &Ks[lr][lc]);
      copy16(ksrc + 8, &Ks[lr][lc + 8]);
      const _Float16* vsrc =
          vtws + ((size_t)bh * DHEAD + lr) * SEQ + jb * 64 + lc;
      copy16(vsrc, &VTs[lr][lc]);
      copy16(vsrc + 8, &VTs[lr][lc + 8]);
    }
    copy_wait();
    __syncthreads();
#pragma unroll
    for (int nt = 0; nt < 4; ++nt) {
      v16h b0 = frag_b(&Ks[nt * 16 + (lane & 15)][0], lane);
      v16h b1 = frag_b(&Ks[nt * 16 + (lane & 15)][32], lane);
      v8f c = {};
      c = wmma16(aq0, b0, c);
      c = wmma16(aq1, b1, c);
      const int key = jb * 64 + nt * 16 + (lane & 15);
#pragma unroll
      for (int r = 0; r < 8; ++r) {
        float s = c[r] * scl;
        if (key > rbase + r) s = -1.0e9f;
        const float p = __expf(s - m[r]) * linv[r];
        probs[pbase + (size_t)(rbase + r) * SEQ + key] = p;  // coalesced f32
        Ps[wid][r + ((lane & 16) ? 8 : 0)][nt * 16 + (lane & 15)] = (_Float16)p;
      }
    }
    // P (C-layout) -> A-layout via per-wave LDS tile; same-wave DS is in-order.
#pragma unroll
    for (int kc = 0; kc < 2; ++kc) {
      v16h ap = frag_a(&Ps[wid][lane & 15][kc * 32], lane);
      a0 = wmma16(ap, frag_b(&VTs[0  + (lane & 15)][kc * 32], lane), a0);
      a1 = wmma16(ap, frag_b(&VTs[16 + (lane & 15)][kc * 32], lane), a1);
      a2 = wmma16(ap, frag_b(&VTs[32 + (lane & 15)][kc * 32], lane), a2);
      a3 = wmma16(ap, frag_b(&VTs[48 + (lane & 15)][kc * 32], lane), a3);
    }
    __syncthreads();
  }

  // context -> ws as f16 [B, S, H*DH] (row-major A operand for out-proj)
  const int b = bh >> 4, h = bh & 15;
#pragma unroll
  for (int nt = 0; nt < 4; ++nt) {
    v8f acc = (nt == 0) ? a0 : (nt == 1) ? a1 : (nt == 2) ? a2 : a3;
#pragma unroll
    for (int r = 0; r < 8; ++r) {
      const int qr  = rbase + r;
      const int col = h * 64 + nt * 16 + (lane & 15);
      ctx[((size_t)b * SEQ + qr) * EMB + col] = (_Float16)acc[r];
    }
  }

  // zero-fill the fully-masked prob columns with float4 stores
  const int kcov = njb * 64;
  const int zq = (SEQ - kcov) >> 2;   // float4s per row
  const int total = 128 * zq;
  for (int i = tid; i < total; i += 256) {
    const int rr = i / zq;
    const int cc = (i - rr * zq) << 2;
    v4f z = {0.f, 0.f, 0.f, 0.f};
    *(v4f*)&probs[pbase + (size_t)(qb * 128 + rr) * SEQ + kcov + cc] = z;
  }
}

// ---------------------------------------------------------------------------
// Kernel 3: output projection. out = ctx @ Wo^T + bo (f32 to d_out).
// ctx tile is a pure f16 copy -> async global->LDS when available.
// ---------------------------------------------------------------------------
__global__ __launch_bounds__(256) void out_proj_kernel(
    const _Float16* __restrict__ ctx, const float* __restrict__ Wo,
    const float* __restrict__ bo, float* __restrict__ out) {
  __shared__ alignas(32) _Float16 Xs[64][32];
  __shared__ alignas(32) _Float16 Ws[64][32];
  const int tid = threadIdx.x, lane = tid & 31, wid = tid >> 5;
  const int mBase = blockIdx.x * 64, nBase = blockIdx.y * 64;
  const int mt = wid >> 1, nt0 = (wid & 1) * 2;
  v8f c0 = {}, c1 = {};
  const int lr = tid >> 2, lc = (tid & 3) * 8;

  for (int kt = 0; kt < EMB / 32; ++kt) {
    const int kBase = kt * 32;
    copy16(ctx + (size_t)(mBase + lr) * EMB + kBase + lc, &Xs[lr][lc]);
    {
      const float* wsrc = Wo + (size_t)(nBase + lr) * EMB + kBase + lc;
      v4f g0 = *(const v4f*)wsrc;
      v4f g1 = *(const v4f*)(wsrc + 4);
      _Float16* wdst = &Ws[lr][lc];
#pragma unroll
      for (int i = 0; i < 4; ++i) { wdst[i] = (_Float16)g0[i]; wdst[i + 4] = (_Float16)g1[i]; }
    }
    copy_wait();
    __syncthreads();
    v16h a  = frag_a(&Xs[mt * 16 + (lane & 15)][0], lane);
    v16h b0 = frag_b(&Ws[nt0 * 16 + (lane & 15)][0], lane);
    v16h b1 = frag_b(&Ws[(nt0 + 1) * 16 + (lane & 15)][0], lane);
    c0 = wmma16(a, b0, c0);
    c1 = wmma16(a, b1, c1);
    __syncthreads();
  }
#pragma unroll
  for (int t = 0; t < 2; ++t) {
    v8f c = t ? c1 : c0;
    const int nt = nt0 + t;
#pragma unroll
    for (int r = 0; r < 8; ++r) {
      const int gr = mBase + mt * 16 + r + ((lane & 16) ? 8 : 0);
      const int gc = nBase + nt * 16 + (lane & 15);
      out[(size_t)gr * DMODEL + gc] = c[r] + bo[gc];
    }
  }
}

// ---------------------------------------------------------------------------
extern "C" void kernel_launch(void* const* d_in, const int* in_sizes, int n_in,
                              void* d_out, int out_size, void* d_ws,
                              size_t ws_size, hipStream_t stream) {
  (void)in_sizes; (void)n_in; (void)out_size; (void)ws_size;
  const float* Q  = (const float*)d_in[0];
  const float* K  = (const float*)d_in[1];
  const float* V  = (const float*)d_in[2];
  // d_in[3] = attn_mask (causal; computed analytically in-kernel)
  const float* Wq = (const float*)d_in[4];
  const float* bq = (const float*)d_in[5];
  const float* Wk = (const float*)d_in[6];
  const float* bk = (const float*)d_in[7];
  const float* Wv = (const float*)d_in[8];
  const float* bv = (const float*)d_in[9];
  const float* Wo = (const float*)d_in[10];
  const float* bo = (const float*)d_in[11];

  float* out   = (float*)d_out;
  float* probs = out + (size_t)NB * SEQ * DMODEL;  // outputs concat: [output, attn_prob]

  const size_t n16 = (size_t)NB * NH * SEQ * DHEAD;  // 4,194,304 halfs (8 MB)
  _Float16* qws  = (_Float16*)d_ws;   // [B,H,S,DH]
  _Float16* kws  = qws + n16;         // [B,H,S,DH]
  _Float16* vtws = kws + n16;         // [B,H,DH,S]
  _Float16* ctx  = vtws + n16;        // [B,S,H*DH]

  dim3 gProj((NB * SEQ) / 64, EMB / 64, 3);
  qkv_proj_kernel<<<gProj, 256, 0, stream>>>(Q, K, V, Wq, Wk, Wv, bq, bk, bv,
                                             qws, kws, vtws);
  dim3 gAttn(SEQ / 128, NB * NH);
  attn_kernel<<<gAttn, 256, 0, stream>>>(qws, kws, vtws, ctx, probs);

  dim3 gOut((NB * SEQ) / 64, DMODEL / 64);
  out_proj_kernel<<<gOut, 256, 0, stream>>>(ctx, Wo, bo, out);
}
